// SimpleGAT_84507776516526
// MI455X (gfx1250) — matrix operand order repaired and checked by reading, not compiled, not run on hardware
//
#include <hip/hip_runtime.h>
#include <hip/hip_bf16.h>
#include <math.h>

// Problem constants (from reference): BS=8, N=2048, DIN=DH=64, H=4
#define GAT_BS   8
#define GAT_N    2048
#define GAT_D    64
#define GAT_H    4

typedef __attribute__((ext_vector_type(16))) _Float16 v16h;
typedef __attribute__((ext_vector_type(8)))  float    v8f;

// ---------------------------------------------------------------------------
// Compute one 16x64 tile of H = X(16x64) @ W^T(64x64) with v_wmma_f32_16x16x32_f16.
// A(16x32 f16) per-lane layout: lanes 0-15 row M=lane hold K=kb+0..15,
// lanes 16-31 same row hold K=kb+16..31.
// B(32x16 f16): lanes 0-15 hold K=kb+0..15 at N=lane; lanes 16-31 hold K=kb+16..31.
// C/D(16x16 f32): VGPR r -> (M=r, N=lane) for lanes 0-15, (M=r+8, N=lane-16) for 16-31.
// acc[t] covers output columns e = 16t + (lane&15).
// ---------------------------------------------------------------------------
__device__ __forceinline__ void wmma_htile(const float* __restrict__ xrow,
                                           const float* __restrict__ W,
                                           int lrow, int hi, v8f acc[4]) {
#pragma unroll
  for (int kb = 0; kb < 64; kb += 32) {
    v16h a;
    const int k0 = kb + (hi ? 16 : 0);
    const int k1 = kb + (hi ? 24 : 8);
#pragma unroll
    for (int i = 0; i < 8; ++i) {
      a[i]     = (_Float16)xrow[k0 + i];
      a[8 + i] = (_Float16)xrow[k1 + i];
    }
#pragma unroll
    for (int t = 0; t < 4; ++t) {
      const float* wr = W + (16 * t + lrow) * 64 + kb + (hi ? 16 : 0);
      v16h bm;
#pragma unroll
      for (int i = 0; i < 16; ++i) bm[i] = (_Float16)wr[i];
      acc[t] = __builtin_amdgcn_wmma_f32_16x16x32_f16(
          /*neg_a=*/false, a, /*neg_b=*/false, bm,
          /*c_mod=*/(short)0, acc[t], /*reuse_a=*/false, /*reuse_b=*/false);
    }
  }
}

// ---------------------------------------------------------------------------
// Pass 1: per (b,head,16-row tile): compute h tile via WMMA, then s_j = h . a2
// s_out layout: [b][head][n]   (BS*H*N floats)
// ---------------------------------------------------------------------------
__global__ __launch_bounds__(32)
void gat_score_kernel(const float* __restrict__ nf, const float* __restrict__ fcw,
                      const float* __restrict__ attnw, float* __restrict__ s_out) {
  const int mtile = blockIdx.x;          // 0..127
  const int head  = blockIdx.y;          // 0..3
  const int b     = blockIdx.z;          // 0..7
  const int lane  = threadIdx.x;         // 0..31
  const int lrow  = lane & 15;
  const int hi    = lane >> 4;
  const int mbase = mtile * 16;

  const float* X = nf + (size_t)b * GAT_N * GAT_D + (size_t)(mbase + lrow) * GAT_D;
  const float* W = fcw + head * GAT_D * GAT_D;

  v8f acc[4] = {};
  wmma_htile(X, W, lrow, hi, acc);

  // s_j for the 16 rows of this tile: s[n] = sum_e h[n,e] * a2[e]
  const float* a2 = attnw + head * (2 * GAT_D) + GAT_D;
  float av[4];
#pragma unroll
  for (int t = 0; t < 4; ++t) av[t] = a2[16 * t + lrow];

  __shared__ float part[32][8];
#pragma unroll
  for (int r = 0; r < 8; ++r) {
    float p = 0.f;
#pragma unroll
    for (int t = 0; t < 4; ++t) p += acc[t][r] * av[t];
    part[lane][r] = p;
  }
  __syncthreads();

  if (lane < 16) {
    const int r    = lane & 7;
    const int base = (lane < 8) ? 0 : 16;       // low lanes own rows mbase+0..7
    float sum = 0.f;
#pragma unroll
    for (int l = 0; l < 16; ++l) sum += part[base + l][r];
    const int row = mbase + ((lane < 8) ? r : (8 + r));
    s_out[((b * GAT_H + head) * GAT_N) + row] = sum;
  }
}

// ---------------------------------------------------------------------------
// Pass 2: softmax over j for each (b,head); overwrite s with normalized weights.
// (softmax_j(s_i + s_j) == softmax_j(s_j): shift invariance)
// ---------------------------------------------------------------------------
__global__ __launch_bounds__(256)
void gat_softmax_kernel(float* __restrict__ s) {
  float* sp = s + (size_t)blockIdx.x * GAT_N;
  __shared__ float red[256];
  const int tid = threadIdx.x;

  float m = -INFINITY;
  for (int n = tid; n < GAT_N; n += 256) m = fmaxf(m, sp[n]);
  red[tid] = m;
  __syncthreads();
  for (int st = 128; st > 0; st >>= 1) {
    if (tid < st) red[tid] = fmaxf(red[tid], red[tid + st]);
    __syncthreads();
  }
  const float gmax = red[0];
  __syncthreads();

  float acc = 0.f;
  for (int n = tid; n < GAT_N; n += 256) acc += expf(sp[n] - gmax);
  red[tid] = acc;
  __syncthreads();
  for (int st = 128; st > 0; st >>= 1) {
    if (tid < st) red[tid] += red[tid + st];
    __syncthreads();
  }
  const float inv = 1.0f / red[0];

  for (int n = tid; n < GAT_N; n += 256) sp[n] = expf(sp[n] - gmax) * inv;
}

// ---------------------------------------------------------------------------
// Pass 3: recompute h tiles (WMMA) and accumulate sum_j w[j]*h[j,:].
// Each block handles 8 tiles (128 rows) for one (b,head); writes 64 partials.
// partial layout: [b][head][group(16)][e(64)]
// ---------------------------------------------------------------------------
__global__ __launch_bounds__(32)
void gat_agg_kernel(const float* __restrict__ nf, const float* __restrict__ fcw,
                    const float* __restrict__ w_s, float* __restrict__ partial) {
  const int g    = blockIdx.x;           // 0..15 (groups of 8 tiles)
  const int head = blockIdx.y;
  const int b    = blockIdx.z;
  const int lane = threadIdx.x;
  const int lrow = lane & 15;
  const int hi   = lane >> 4;

  const float* Xb = nf + (size_t)b * GAT_N * GAT_D;
  const float* W  = fcw + head * GAT_D * GAT_D;
  const float* wrow = w_s + (size_t)(b * GAT_H + head) * GAT_N;

  float csum[4] = {0.f, 0.f, 0.f, 0.f};
  for (int mt = 0; mt < 8; ++mt) {
    const int mbase = (g * 8 + mt) * 16;
    v8f acc[4] = {};
    wmma_htile(Xb + (size_t)(mbase + lrow) * GAT_D, W, lrow, hi, acc);
#pragma unroll
    for (int r = 0; r < 8; ++r) {
      const float wv = wrow[mbase + r + (hi ? 8 : 0)];
#pragma unroll
      for (int t = 0; t < 4; ++t) csum[t] += acc[t][r] * wv;
    }
  }

  __shared__ float part[32][4];
#pragma unroll
  for (int t = 0; t < 4; ++t) part[lane][t] = csum[t];
  __syncthreads();

  if (lane < 16) {
    float* pp = partial + (size_t)(((b * GAT_H + head) * 16 + g)) * GAT_D;
#pragma unroll
    for (int t = 0; t < 4; ++t)
      pp[16 * t + lane] = part[lane][t] + part[lane + 16][t];
  }
}

// ---------------------------------------------------------------------------
// Pass 4: deterministic reduce of the 64 (=H*16) partials per batch.
// ---------------------------------------------------------------------------
__global__ __launch_bounds__(64)
void gat_reduce_kernel(const float* __restrict__ partial, float* __restrict__ vacc) {
  const int b = blockIdx.x;
  const int e = threadIdx.x;              // 0..63
  const float* p = partial + (size_t)b * GAT_H * 16 * GAT_D;
  float s = 0.f;
  for (int i = 0; i < GAT_H * 16; ++i) s += p[i * GAT_D + e];
  vacc[b * GAT_D + e] = s;
}

// ---------------------------------------------------------------------------
// Pass 5: emb_row[b,:] = vacc[b] @ merge_w^T + merge_b; broadcast to all N rows.
// grid (BS, 32 chunks of 64 rows), 256 threads.
// ---------------------------------------------------------------------------
__global__ __launch_bounds__(256)
void gat_merge_bcast_kernel(const float* __restrict__ vacc,
                            const float* __restrict__ mw,
                            const float* __restrict__ mb,
                            float* __restrict__ out) {
  const int b     = blockIdx.x;
  const int chunk = blockIdx.y;           // 0..31 -> rows chunk*64 .. +63
  const int tid   = threadIdx.x;

  __shared__ float v[GAT_D];
  __shared__ float row[GAT_D];
  if (tid < GAT_D) v[tid] = vacc[b * GAT_D + tid];
  __syncthreads();
  if (tid < GAT_D) {
    float acc = mb[tid];
#pragma unroll
    for (int k = 0; k < GAT_D; ++k) acc += v[k] * mw[tid * GAT_D + k];
    row[tid] = acc;
  }
  __syncthreads();

  float* op = out + ((size_t)b * GAT_N + (size_t)chunk * 64) * GAT_D;
  // 64 rows x 64 cols = 4096 floats = 1024 float4 stores
  for (int i = tid; i < 1024; i += 256) {
    const int c = (i * 4) & 63;
    float4 vv = make_float4(row[c], row[c + 1], row[c + 2], row[c + 3]);
    ((float4*)op)[i] = vv;
  }
}

// ---------------------------------------------------------------------------
extern "C" void kernel_launch(void* const* d_in, const int* in_sizes, int n_in,
                              void* d_out, int out_size, void* d_ws, size_t ws_size,
                              hipStream_t stream) {
  const float* nf    = (const float*)d_in[0];   // (8,2048,64)
  // d_in[1] edge_features: unused by the reference computation
  const float* fcw   = (const float*)d_in[2];   // (4,64,64)
  const float* attnw = (const float*)d_in[3];   // (4,128)
  const float* mw    = (const float*)d_in[4];   // (64,64)
  const float* mb    = (const float*)d_in[5];   // (64,)
  float* out = (float*)d_out;                   // (8,2048,64)

  // Workspace layout (floats): s[8*4*2048] | partial[8*4*16*64] | vacc[8*64]
  float* ws      = (float*)d_ws;
  float* s       = ws;                                       // 65536 floats
  float* partial = s + GAT_BS * GAT_H * GAT_N;               // 32768 floats
  float* vacc    = partial + GAT_BS * GAT_H * 16 * GAT_D;    //   512 floats

  gat_score_kernel<<<dim3(GAT_N / 16, GAT_H, GAT_BS), 32, 0, stream>>>(nf, fcw, attnw, s);
  gat_softmax_kernel<<<GAT_BS * GAT_H, 256, 0, stream>>>(s);
  gat_agg_kernel<<<dim3(16, GAT_H, GAT_BS), 32, 0, stream>>>(nf, fcw, s, partial);
  gat_reduce_kernel<<<GAT_BS, 64, 0, stream>>>(partial, vacc);
  gat_merge_bcast_kernel<<<dim3(GAT_BS, 32), 256, 0, stream>>>(vacc, mw, mb, out);
}
